// RND1DecoderLayer_22668837388635
// MI455X (gfx1250) — compile-verified, matrix operand order checked by reading, not run on hardware
//
#include <hip/hip_runtime.h>
#include <hip/hip_bf16.h>
#include <cstdint>

#define H_DIM 1024
#define NHEAD 16
#define KVHEAD 4
#define HD 64
#define NEXP 16
#define TOPK 2
#define FDIM 1024
#define BATCH 2
#define SEQ 1024
#define TTOK (BATCH*SEQ)
#define CAP 4352            /* padded pair capacity: 4096 pairs + <=15 pad/expert, 16-aligned */
#define MTILES (CAP/16)

typedef unsigned short u16;
typedef __attribute__((ext_vector_type(8)))  unsigned short u16x8;
typedef __attribute__((ext_vector_type(8)))  __bf16         bf16x8v;
typedef __attribute__((ext_vector_type(16))) __bf16         v16bf;
typedef __attribute__((ext_vector_type(8)))  float          v8f;

union Frag { v16bf v; u16x8 h[2]; };

// hardware RNE f32->bf16 (v_cvt)
__device__ __forceinline__ u16 f2bf(float f){
  return __builtin_bit_cast(u16, (__bf16)f);
}

__device__ __forceinline__ u16x8 cvt8(float4 a, float4 b){
  bf16x8v r;
  r[0]=(__bf16)a.x; r[1]=(__bf16)a.y; r[2]=(__bf16)a.z; r[3]=(__bf16)a.w;
  r[4]=(__bf16)b.x; r[5]=(__bf16)b.y; r[6]=(__bf16)b.z; r[7]=(__bf16)b.w;
  return __builtin_bit_cast(u16x8, r);
}

__device__ __forceinline__ v8f wmma_bf16(v16bf a, v16bf b, v8f c){
  return __builtin_amdgcn_wmma_f32_16x16x32_bf16(false, a, false, b, (short)0, c, false, false);
}

__device__ __forceinline__ Frag load_bfrag_f32(const float* __restrict__ bp){
  Frag b;
  b.h[0] = cvt8(*(const float4*)bp,        *(const float4*)(bp + 4));
  b.h[1] = cvt8(*(const float4*)(bp + 16), *(const float4*)(bp + 20));
  return b;
}

// -------- cross-lane reductions via v_permlane16 (VALU, no LDS traffic) ----
__device__ __forceinline__ float pl16(float v, unsigned lo, unsigned hi){
  return __builtin_bit_cast(float, __builtin_amdgcn_permlane16(
      __builtin_bit_cast(int, v), __builtin_bit_cast(int, v),
      (int)lo, (int)hi, false, false));
}
__device__ __forceinline__ float plx16(float v){   // identity swap across 16-lane halves
  return __builtin_bit_cast(float, __builtin_amdgcn_permlanex16(
      __builtin_bit_cast(int, v), __builtin_bit_cast(int, v),
      (int)0x76543210, (int)0xFEDCBA98, false, false));
}
// max over each 16-lane group (xor1/2/4/8 butterfly)
__device__ __forceinline__ float red_max16(float v){
  v = fmaxf(v, pl16(v, 0x67452301u, 0xEFCDAB89u));
  v = fmaxf(v, pl16(v, 0x54761032u, 0xDCFE98BAu));
  v = fmaxf(v, pl16(v, 0x32107654u, 0xBA98FEDCu));
  v = fmaxf(v, pl16(v, 0xFEDCBA98u, 0x76543210u));
  return v;
}
// sum over all 32 lanes
__device__ __forceinline__ float red_sum32(float v){
  v += pl16(v, 0x67452301u, 0xEFCDAB89u);
  v += pl16(v, 0x54761032u, 0xDCFE98BAu);
  v += pl16(v, 0x32107654u, 0xBA98FEDCu);
  v += pl16(v, 0xFEDCBA98u, 0x76543210u);
  v += plx16(v);
  return v;
}

// ---------------------------------------------------------------- rmsnorm
__global__ __launch_bounds__(256) void rmsnorm_bf16_k(const float* __restrict__ x,
                                                      const float* __restrict__ w,
                                                      u16* __restrict__ out){
  int t = blockIdx.x;
  const float* xr = x + (size_t)t * H_DIM;
  float s = 0.f;
  for (int i = threadIdx.x; i < H_DIM; i += 256){ float v = xr[i]; s += v*v; }
  s = red_sum32(s);
  __shared__ float red[8];
  if ((threadIdx.x & 31) == 0) red[threadIdx.x >> 5] = s;
  __syncthreads();
  float tot = 0.f;
  #pragma unroll
  for (int j = 0; j < 8; ++j) tot += red[j];
  float inv = rsqrtf(tot / (float)H_DIM + 1e-6f);
  u16* orow = out + (size_t)t * H_DIM;
  for (int i = threadIdx.x; i < H_DIM; i += 256) orow[i] = f2bf(xr[i] * inv * w[i]);
}

// ------------------------------------------------ strip GEMM (wave = 16 x 16*NSUB)
// C[M,N] = A_bf16[M,K] @ W_f32[N,K]^T (+resid).  A fragment shared by NSUB wmmas.
template<int NSUB, bool RESID>
__global__ __launch_bounds__(256) void gemm_bf16w_k(const u16* __restrict__ A, int lda,
                                                    const float* __restrict__ W, int ldw,
                                                    float* __restrict__ C, int ldc,
                                                    const float* __restrict__ resid,
                                                    int Mt, int Nstrips, int K){
  int wave = (blockIdx.x * 256 + threadIdx.x) >> 5;
  if (wave >= Mt * Nstrips) return;                  // wave-uniform
  int lane = threadIdx.x & 31, hf = lane >> 4, ln = lane & 15;
  int mt = wave / Nstrips, ns = wave % Nstrips;
  const u16* Ar = A + (size_t)(mt*16 + ln) * lda;
  const float* Br[NSUB];
  v8f c[NSUB];
  #pragma unroll
  for (int s = 0; s < NSUB; ++s){
    Br[s] = W + (size_t)((unsigned)ns*(16u*NSUB) + s*16 + ln) * ldw;
    c[s] = (v8f){};
  }
  for (int k0 = 0; k0 < K; k0 += 32){
    Frag a;
    a.h[0] = *(const u16x8*)(Ar + k0 + 8*hf);
    a.h[1] = *(const u16x8*)(Ar + k0 + 16 + 8*hf);
    #pragma unroll
    for (int s = 0; s < NSUB; ++s){
      Frag b = load_bfrag_f32(Br[s] + k0 + 8*hf);
      c[s] = wmma_bf16(a.v, b.v, c[s]);
    }
  }
  unsigned colbase = (unsigned)ns*(16u*NSUB) + ln;
  #pragma unroll
  for (int r = 0; r < 8; ++r){
    size_t row = (size_t)((unsigned)mt*16u + r + 8u*hf);
    float* crow = C + row * (unsigned)ldc + colbase;
    const float* rrow = RESID ? (resid + row * (unsigned)ldc + colbase) : nullptr;
    #pragma unroll
    for (int s = 0; s < NSUB; ++s){
      float x = c[s][r];
      if constexpr (RESID) x += rrow[s*16];
      crow[s*16] = x;
    }
  }
}

// ------------------------------------- per-head RMSNorm + RoPE (wave/row)
__global__ __launch_bounds__(256) void qk_norm_rope_k(const float* __restrict__ src, int nh,
                                                      const float* __restrict__ nw,
                                                      const int* __restrict__ pos_ids,
                                                      float scale, u16* __restrict__ dst){
  int row  = (blockIdx.x * 256 + threadIdx.x) >> 5;
  int lane = threadIdx.x & 31;
  int t = row / nh, head = row % nh;
  int b = t / SEQ, s = t % SEQ;
  const float* xr = src + (size_t)t * nh * HD + head * HD;
  float x0 = xr[lane], x1 = xr[lane + 32];
  float ss = red_sum32(x0*x0 + x1*x1);
  float inv = rsqrtf(ss / (float)HD + 1e-6f);
  float n0 = x0 * inv * nw[lane], n1 = x1 * inv * nw[lane + 32];
  float pos = (float)pos_ids[t];
  float f = pos * __expf(-(float)lane * (1.0f/32.0f) * logf(10000.0f));
  float sn, c;
  __sincosf(f, &sn, &c);
  float y0 = n0*c - n1*sn;           // rotate_half: d<32 -> -x[d+32]
  float y1 = n1*c + n0*sn;           //              d>=32 -> x[d-32]
  u16* d = dst + ((size_t)(b*nh + head) * SEQ + s) * HD;
  d[lane]      = f2bf(y0 * scale);
  d[lane + 32] = f2bf(y1 * scale);
}

// -------------------------------------------- V -> V^T (bf16) for PV wmma
__global__ __launch_bounds__(256) void v_transpose_k(const float* __restrict__ v,
                                                     u16* __restrict__ vt){
  int idx = blockIdx.x * 256 + threadIdx.x;          // over TTOK*KVHEAD*HD
  int d  = idx & (HD - 1);
  int kv = (idx >> 6) & (KVHEAD - 1);
  int t  = idx >> 8;
  int b = t / SEQ, s = t % SEQ;
  float val = v[(size_t)t * (KVHEAD*HD) + kv*HD + d];
  vt[((size_t)(b*KVHEAD + kv) * HD + d) * SEQ + s] = f2bf(val);
}

// ------------------------------------------------ flash attention (wave =
// one (b,h,16-row q tile); 32-key inner tiles; online softmax)
__global__ __launch_bounds__(256) void flash_attn_k(const u16* __restrict__ qb,
                                                    const u16* __restrict__ kb,
                                                    const u16* __restrict__ vtb,
                                                    u16* __restrict__ attnb){
  __shared__ __align__(16) u16 plds[8][16*32];
  int wave = (blockIdx.x * 256 + threadIdx.x) >> 5;
  int lane = threadIdx.x & 31;
  int hf = lane >> 4, ln = lane & 15;
  int qt = wave & 63;                 // S/16 = 64 q tiles
  int bh = wave >> 6;
  int h  = bh & (NHEAD - 1);
  int b  = bh >> 4;
  int kvh = h >> 2;                   // GQA: NH/KV = 4
  const u16* qrow  = qb  + ((size_t)(b*NHEAD  + h  ) * SEQ + qt*16 + ln) * HD;
  const u16* kbase = kb  +  (size_t)(b*KVHEAD + kvh) * SEQ * HD;
  const u16* vbase = vtb +  (size_t)(b*KVHEAD + kvh) * HD * SEQ;

  Frag qa0, qa1;                      // q is pre-scaled by 1/sqrt(HD)
  qa0.h[0] = *(const u16x8*)(qrow      + 8*hf);
  qa0.h[1] = *(const u16x8*)(qrow + 16 + 8*hf);
  qa1.h[0] = *(const u16x8*)(qrow + 32 + 8*hf);
  qa1.h[1] = *(const u16x8*)(qrow + 48 + 8*hf);

  Frag ones;                          // B = all-ones: P x 1 -> broadcast row sums
  #pragma unroll
  for (int e = 0; e < 8; ++e){ ones.h[0][e] = 0x3F80; ones.h[1][e] = 0x3F80; }

  v8f O0 = {}, O1 = {}, O2 = {}, O3 = {};
  float m[8], lsum[8];
  #pragma unroll
  for (int r = 0; r < 8; ++r){ m[r] = -1e30f; lsum[r] = 0.f; }
  u16* pl = &plds[threadIdx.x >> 5][0];

  for (int j0 = 0; j0 < SEQ; j0 += 32){
    v8f s0 = {}, s1 = {};
    {
      const u16* kp = kbase + (size_t)(j0 + ln) * HD;
      Frag kf;
      kf.h[0] = *(const u16x8*)(kp      + 8*hf);
      kf.h[1] = *(const u16x8*)(kp + 16 + 8*hf);
      s0 = wmma_bf16(qa0.v, kf.v, s0);
      kf.h[0] = *(const u16x8*)(kp + 32 + 8*hf);
      kf.h[1] = *(const u16x8*)(kp + 48 + 8*hf);
      s0 = wmma_bf16(qa1.v, kf.v, s0);
    }
    {
      const u16* kp = kbase + (size_t)(j0 + 16 + ln) * HD;
      Frag kf;
      kf.h[0] = *(const u16x8*)(kp      + 8*hf);
      kf.h[1] = *(const u16x8*)(kp + 16 + 8*hf);
      s1 = wmma_bf16(qa0.v, kf.v, s1);
      kf.h[0] = *(const u16x8*)(kp + 32 + 8*hf);
      kf.h[1] = *(const u16x8*)(kp + 48 + 8*hf);
      s1 = wmma_bf16(qa1.v, kf.v, s1);
    }
    // online softmax: row max via permlane butterfly (VALU only)
    float alpha[8];
    #pragma unroll
    for (int r = 0; r < 8; ++r){
      float mx = red_max16(fmaxf(s0[r], s1[r]));
      float mn = fmaxf(m[r], mx);
      alpha[r] = __expf(m[r] - mn);
      m[r] = mn;
      s0[r] = __expf(s0[r] - mn);
      s1[r] = __expf(s1[r] - mn);
    }
    #pragma unroll
    for (int r = 0; r < 8; ++r){
      O0[r] *= alpha[r]; O1[r] *= alpha[r]; O2[r] *= alpha[r]; O3[r] *= alpha[r];
    }
    // P (C-layout) -> A-layout via per-wave LDS patch
    #pragma unroll
    for (int r = 0; r < 8; ++r){
      int rr = r + 8*hf;
      pl[rr*32 + ln]      = f2bf(s0[r]);
      pl[rr*32 + 16 + ln] = f2bf(s1[r]);
    }
    asm volatile("s_wait_dscnt 0x0" ::: "memory");
    Frag pa;
    pa.h[0] = *(const u16x8*)(pl + ln*32      + 8*hf);
    pa.h[1] = *(const u16x8*)(pl + ln*32 + 16 + 8*hf);
    // row sums via one extra wmma: (P x ones) broadcasts rowsum to all lanes
    v8f rs = {};
    rs = wmma_bf16(pa.v, ones.v, rs);
    #pragma unroll
    for (int r = 0; r < 8; ++r) lsum[r] = lsum[r] * alpha[r] + rs[r];
    #pragma unroll
    for (int sub = 0; sub < 4; ++sub){
      const u16* vp = vbase + (size_t)(sub*16 + ln) * SEQ + j0;
      Frag vf;
      vf.h[0] = *(const u16x8*)(vp      + 8*hf);
      vf.h[1] = *(const u16x8*)(vp + 16 + 8*hf);
      v8f& O = (sub == 0) ? O0 : (sub == 1) ? O1 : (sub == 2) ? O2 : O3;
      O = wmma_bf16(pa.v, vf.v, O);
    }
  }
  #pragma unroll
  for (int r = 0; r < 8; ++r){
    float inv = 1.0f / lsum[r];
    int t = b*SEQ + qt*16 + r + 8*hf;
    u16* op = attnb + (size_t)t * (NHEAD*HD) + h*HD + ln;
    op[0]  = f2bf(O0[r] * inv);
    op[16] = f2bf(O1[r] * inv);
    op[32] = f2bf(O2[r] * inv);
    op[48] = f2bf(O3[r] * inv);
  }
}

// ------------------------------------------------------------- MoE routing
__global__ void route_init_k(int* counts, int* cursor){
  int i = threadIdx.x;
  if (i < NEXP){ counts[i] = 0; cursor[i] = 0; }
}

__global__ __launch_bounds__(256) void route_topk_k(const float* __restrict__ logits,
                                                    int* __restrict__ sel,
                                                    float* __restrict__ wgt,
                                                    int* __restrict__ counts){
  int t = blockIdx.x * 256 + threadIdx.x;
  if (t >= TTOK) return;
  const float* lr = logits + (size_t)t * NEXP;
  float p[NEXP];
  float mx = lr[0];
  #pragma unroll
  for (int i = 1; i < NEXP; ++i) mx = fmaxf(mx, lr[i]);
  #pragma unroll
  for (int i = 0; i < NEXP; ++i) p[i] = __expf(lr[i] - mx);
  int i0 = 0; float b0 = p[0];
  #pragma unroll
  for (int i = 1; i < NEXP; ++i) if (p[i] > b0){ b0 = p[i]; i0 = i; }
  int i1 = -1; float b1 = -1.f;
  #pragma unroll
  for (int i = 0; i < NEXP; ++i) if (i != i0 && p[i] > b1){ b1 = p[i]; i1 = i; }
  float inv = 1.0f / (b0 + b1);              // softmax denom cancels in norm
  sel[2*t] = i0;  sel[2*t+1] = i1;
  wgt[2*t] = b0 * inv; wgt[2*t+1] = b1 * inv;
  atomicAdd(&counts[i0], 1);
  atomicAdd(&counts[i1], 1);
}

__global__ void route_scan_k(const int* __restrict__ counts, int* __restrict__ offs){
  if (threadIdx.x == 0){
    int o = 0;
    for (int e = 0; e < NEXP; ++e){ offs[e] = o; o += (counts[e] + 15) & ~15; }
    offs[NEXP] = o;
  }
}

__global__ __launch_bounds__(256) void route_fill_k(const int* __restrict__ offs,
                                                    int* __restrict__ pair_token,
                                                    float* __restrict__ wgt_slot,
                                                    int* __restrict__ expert_slot){
  int slot = blockIdx.x * 256 + threadIdx.x;
  if (slot >= CAP) return;
  int e = NEXP - 1;
  #pragma unroll
  for (int i = NEXP - 1; i >= 0; --i) if (slot < offs[i+1]) e = i;
  pair_token[slot] = 0;
  wgt_slot[slot]   = 0.f;
  expert_slot[slot] = e;
}

__global__ __launch_bounds__(256) void route_scatter_k(const int* __restrict__ sel,
                                                       const float* __restrict__ wgt,
                                                       const int* __restrict__ offs,
                                                       int* __restrict__ cursor,
                                                       int* __restrict__ pair_token,
                                                       float* __restrict__ wgt_slot,
                                                       int* __restrict__ pos_of){
  int t = blockIdx.x * 256 + threadIdx.x;
  if (t >= TTOK) return;
  #pragma unroll
  for (int k = 0; k < TOPK; ++k){
    int e = sel[2*t + k];
    int pos = offs[e] + atomicAdd(&cursor[e], 1);
    pair_token[pos] = t;
    wgt_slot[pos]   = wgt[2*t + k];
    pos_of[2*t + k] = pos;
  }
}

// ----------------- fused gate/up GEMM + SiLU epilogue (wave = 16 x 32 F-cols)
__global__ __launch_bounds__(256) void moe_gateup_k(const u16* __restrict__ X,
                                                    const float* __restrict__ Wg,
                                                    const float* __restrict__ Wu,
                                                    const int* __restrict__ pair_token,
                                                    const int* __restrict__ expert_slot,
                                                    u16* __restrict__ Abuf){
  const int NS = 2;                       // two 16-col subtiles per wave
  const int Nstrips = FDIM / (16*NS);     // 32
  int wave = (blockIdx.x * 256 + threadIdx.x) >> 5;
  if (wave >= MTILES * Nstrips) return;
  int lane = threadIdx.x & 31, hf = lane >> 4, ln = lane & 15;
  int mt = wave / Nstrips, ns = wave % Nstrips;
  int e   = expert_slot[mt*16];
  int tok = pair_token[mt*16 + ln];
  const u16* Ar = X + (size_t)tok * H_DIM;
  const float* Bg[NS];
  const float* Bu[NS];
  v8f g[NS], u[NS];
  #pragma unroll
  for (int s = 0; s < NS; ++s){
    size_t fcol = (size_t)((unsigned)ns*(16u*NS) + s*16 + ln);
    Bg[s] = Wg + ((size_t)e * FDIM + fcol) * H_DIM;
    Bu[s] = Wu + ((size_t)e * FDIM + fcol) * H_DIM;
    g[s] = (v8f){}; u[s] = (v8f){};
  }
  for (int k0 = 0; k0 < H_DIM; k0 += 32){
    Frag a;
    a.h[0] = *(const u16x8*)(Ar + k0 + 8*hf);
    a.h[1] = *(const u16x8*)(Ar + k0 + 16 + 8*hf);
    #pragma unroll
    for (int s = 0; s < NS; ++s){
      Frag bg = load_bfrag_f32(Bg[s] + k0 + 8*hf);
      g[s] = wmma_bf16(a.v, bg.v, g[s]);
      Frag bu = load_bfrag_f32(Bu[s] + k0 + 8*hf);
      u[s] = wmma_bf16(a.v, bu.v, u[s]);
    }
  }
  #pragma unroll
  for (int r = 0; r < 8; ++r){
    u16* arow = Abuf + (size_t)((unsigned)mt*16u + r + 8u*hf) * FDIM
                     + (unsigned)ns*(16u*NS) + ln;
    #pragma unroll
    for (int s = 0; s < NS; ++s){
      float gv = g[s][r];
      float av = (gv / (1.0f + __expf(-gv))) * u[s][r];   // silu(g)*u
      arow[s*16] = f2bf(av);
    }
  }
}

// ------------- down GEMM + routing-weight scale epilogue (wave = 16 x 64)
__global__ __launch_bounds__(256) void moe_down_k(const u16* __restrict__ Abuf,
                                                  const float* __restrict__ Wd,
                                                  const int* __restrict__ expert_slot,
                                                  const float* __restrict__ wgt_slot,
                                                  float* __restrict__ Ypair){
  const int NS = 4;
  const int Nstrips = H_DIM / (16*NS);    // 16
  int wave = (blockIdx.x * 256 + threadIdx.x) >> 5;
  if (wave >= MTILES * Nstrips) return;
  int lane = threadIdx.x & 31, hf = lane >> 4, ln = lane & 15;
  int mt = wave / Nstrips, ns = wave % Nstrips;
  int e = expert_slot[mt*16];
  const u16* Ar = Abuf + (size_t)(mt*16 + ln) * FDIM;
  const float* Br[NS];
  v8f c[NS];
  #pragma unroll
  for (int s = 0; s < NS; ++s){
    Br[s] = Wd + ((size_t)e * H_DIM + (unsigned)ns*(16u*NS) + s*16 + ln) * FDIM;
    c[s] = (v8f){};
  }
  for (int k0 = 0; k0 < FDIM; k0 += 32){
    Frag a;
    a.h[0] = *(const u16x8*)(Ar + k0 + 8*hf);
    a.h[1] = *(const u16x8*)(Ar + k0 + 16 + 8*hf);
    #pragma unroll
    for (int s = 0; s < NS; ++s){
      Frag b = load_bfrag_f32(Br[s] + k0 + 8*hf);
      c[s] = wmma_bf16(a.v, b.v, c[s]);
    }
  }
  #pragma unroll
  for (int r = 0; r < 8; ++r){
    unsigned row = (unsigned)mt*16u + r + 8u*hf;
    float wr = wgt_slot[row];
    float* yrow = Ypair + (size_t)row * H_DIM + (unsigned)ns*(16u*NS) + ln;
    #pragma unroll
    for (int s = 0; s < NS; ++s) yrow[s*16] = c[s][r] * wr;
  }
}

// ---------------------------------------- deterministic 2-way combine
__global__ __launch_bounds__(256) void combine_k(const float* __restrict__ x1,
                                                 const float* __restrict__ Ypair,
                                                 const int* __restrict__ pos_of,
                                                 float* __restrict__ out){
  int idx = blockIdx.x * 256 + threadIdx.x;
  int t = idx >> 10;                   // H = 1024
  int hcol = idx & 1023;
  int p0 = pos_of[2*t], p1 = pos_of[2*t + 1];
  out[idx] = x1[idx] + Ypair[(size_t)p0 * H_DIM + hcol]
                     + Ypair[(size_t)p1 * H_DIM + hcol];
}

// ------------------------------------------------------------------ launch
extern "C" void kernel_launch(void* const* d_in, const int* in_sizes, int n_in,
                              void* d_out, int out_size, void* d_ws, size_t ws_size,
                              hipStream_t stream){
  const float* hidden  = (const float*)d_in[0];
  const int*   pos_ids = (const int*)  d_in[1];
  const float* ln1_w   = (const float*)d_in[2];
  const float* ln2_w   = (const float*)d_in[3];
  const float* q_w     = (const float*)d_in[4];
  const float* k_w     = (const float*)d_in[5];
  const float* v_w     = (const float*)d_in[6];
  const float* o_w     = (const float*)d_in[7];
  const float* qn_w    = (const float*)d_in[8];
  const float* kn_w    = (const float*)d_in[9];
  const float* gate_w  = (const float*)d_in[10];
  const float* w_gate  = (const float*)d_in[11];
  const float* w_up    = (const float*)d_in[12];
  const float* w_down  = (const float*)d_in[13];

  float* out        = (float*)d_out;
  float* out_logits = out + (size_t)TTOK * H_DIM;   // tuple output 2

  char* wp = (char*)d_ws;
  auto alloc = [&](size_t bytes) -> void* {
    void* p = wp; wp += (bytes + 255) & ~(size_t)255; return p;
  };
  u16*   h1b   = (u16*)  alloc((size_t)TTOK * H_DIM * 2);
  float* qf    = (float*)alloc((size_t)TTOK * NHEAD  * HD * 4);
  float* kf    = (float*)alloc((size_t)TTOK * KVHEAD * HD * 4);
  float* vf    = (float*)alloc((size_t)TTOK * KVHEAD * HD * 4);
  u16*   qbf   = (u16*)  alloc((size_t)TTOK * NHEAD  * HD * 2);
  u16*   kbf   = (u16*)  alloc((size_t)TTOK * KVHEAD * HD * 2);
  u16*   vtb   = (u16*)  alloc((size_t)TTOK * KVHEAD * HD * 2);
  u16*   attnb = (u16*)  alloc((size_t)TTOK * NHEAD  * HD * 2);
  float* x1    = (float*)alloc((size_t)TTOK * H_DIM * 4);
  u16*   h2b   = (u16*)  alloc((size_t)TTOK * H_DIM * 2);
  int*   sel   = (int*)  alloc(TTOK * 2 * 4);
  float* wgt   = (float*)alloc(TTOK * 2 * 4);
  int*   counts = (int*) alloc(64);
  int*   cursor = (int*) alloc(64);
  int*   offs   = (int*) alloc((NEXP + 1) * 4);
  int*   pos_of = (int*) alloc(TTOK * 2 * 4);
  int*   pair_token  = (int*)  alloc(CAP * 4);
  float* wgt_slot    = (float*)alloc(CAP * 4);
  int*   expert_slot = (int*)  alloc(CAP * 4);
  u16*   abuf  = (u16*)  alloc((size_t)CAP * FDIM * 2);
  float* ypair = (float*)alloc((size_t)CAP * H_DIM * 4);

  // ---- attention ----
  rmsnorm_bf16_k<<<TTOK, 256, 0, stream>>>(hidden, ln1_w, h1b);
  gemm_bf16w_k<4,false><<<(128*16)/8, 256, 0, stream>>>(h1b, H_DIM, q_w, H_DIM, qf, NHEAD*HD,
                                               nullptr, TTOK/16, (NHEAD*HD)/64, H_DIM);
  gemm_bf16w_k<4,false><<<(128*4)/8, 256, 0, stream>>>(h1b, H_DIM, k_w, H_DIM, kf, KVHEAD*HD,
                                               nullptr, TTOK/16, (KVHEAD*HD)/64, H_DIM);
  gemm_bf16w_k<4,false><<<(128*4)/8, 256, 0, stream>>>(h1b, H_DIM, v_w, H_DIM, vf, KVHEAD*HD,
                                               nullptr, TTOK/16, (KVHEAD*HD)/64, H_DIM);
  qk_norm_rope_k<<<(TTOK*NHEAD)/8,  256, 0, stream>>>(qf, NHEAD,  qn_w, pos_ids, 0.125f, qbf);
  qk_norm_rope_k<<<(TTOK*KVHEAD)/8, 256, 0, stream>>>(kf, KVHEAD, kn_w, pos_ids, 1.0f,   kbf);
  v_transpose_k<<<(TTOK*KVHEAD*HD)/256, 256, 0, stream>>>(vf, vtb);
  flash_attn_k<<<(BATCH*NHEAD*(SEQ/16))/8, 256, 0, stream>>>(qbf, kbf, vtb, attnb);
  gemm_bf16w_k<4,true><<<(128*16)/8, 256, 0, stream>>>(attnb, NHEAD*HD, o_w, NHEAD*HD, x1, H_DIM,
                                               hidden, TTOK/16, H_DIM/64, NHEAD*HD);

  // ---- MoE ----
  rmsnorm_bf16_k<<<TTOK, 256, 0, stream>>>(x1, ln2_w, h2b);
  gemm_bf16w_k<1,false><<<(128*1)/8, 256, 0, stream>>>(h2b, H_DIM, gate_w, H_DIM, out_logits,
                                               NEXP, nullptr, TTOK/16, NEXP/16, H_DIM);
  route_init_k<<<1, 64, 0, stream>>>(counts, cursor);
  route_topk_k<<<TTOK/256, 256, 0, stream>>>(out_logits, sel, wgt, counts);
  route_scan_k<<<1, 32, 0, stream>>>(counts, offs);
  route_fill_k<<<(CAP + 255)/256, 256, 0, stream>>>(offs, pair_token, wgt_slot, expert_slot);
  route_scatter_k<<<TTOK/256, 256, 0, stream>>>(sel, wgt, offs, cursor,
                                                pair_token, wgt_slot, pos_of);
  moe_gateup_k<<<(MTILES*(FDIM/32))/8, 256, 0, stream>>>(h2b, w_gate, w_up,
                                                         pair_token, expert_slot, abuf);
  moe_down_k<<<(MTILES*(H_DIM/64))/8, 256, 0, stream>>>(abuf, w_down, expert_slot,
                                                        wgt_slot, ypair);
  combine_k<<<(TTOK*H_DIM)/256, 256, 0, stream>>>(x1, ypair, pos_of, out);
}
